// Sound_Classification_CNN_21904333209752
// MI455X (gfx1250) — 8-bit matrix operand order repaired and checked by reading, compiled, not run on hardware
//
#include <hip/hip_runtime.h>
#include <hip/hip_bf16.h>

typedef __attribute__((ext_vector_type(8))) int v8i;

union AFrag { v8i v; int2 d[4]; };
union BFrag { v8i v; int4 q[2]; };

#define OFF_WB1 0
#define OFF_WB2 512
#define OFF_WBO 20992
#define OFF_WBL 26112
#define OFF_H1B 1946112
#define OFF_H2B 76919296
#define OFF_H3B 108376576

__device__ __forceinline__ int fsign(float x) { return x > 0.f ? 1 : (x < 0.f ? -1 : 0); }

__global__ void prep_kernel(const float* c1w, const float* c2w, const float* linw,
                            const float* outw, signed char* wb1, signed char* wb2,
                            signed char* wbo, signed char* wbl) {
    int i = blockIdx.x * blockDim.x + threadIdx.x;
    if (i < 288) {
        wb1[i] = (signed char)fsign(c1w[i]);
    } else if (i < 288 + 20480) {
        int j = i - 288;
        int o = j / 320, k = j % 320;
        signed char s = 0;
        if (k < 288) {
            int tap = k >> 5, c = k & 31;
            int kh = tap / 3, kw = tap % 3;
            s = (signed char)fsign(c2w[((o * 32 + c) * 3 + kh) * 3 + kw]);
        }
        wb2[j] = s;
    } else if (i < 288 + 20480 + 5000) {
        int j = i - (288 + 20480);
        wbo[j] = (signed char)fsign(outw[j]);
    } else if (i < 288 + 20480 + 5000 + 1920000) {
        int j = i - (288 + 20480 + 5000);
        wbl[j] = (signed char)fsign(linw[j]);
    }
}

__global__ void conv1_kernel(const float* __restrict__ x, const signed char* __restrict__ wb1,
                             const float* __restrict__ b1, const float* __restrict__ g1,
                             const float* __restrict__ be1, const float* __restrict__ mu1,
                             const float* __restrict__ var1, signed char* __restrict__ h1b) {
    int idx = blockIdx.x * blockDim.x + threadIdx.x;
    if (idx >= 8192 * 11 * 26) return;
    int ow = idx % 26, t = idx / 26;
    int oh = t % 11, b = t / 11;
    const float* xp = x + (b * 24 + oh * 2) * 54 + ow * 2;
    int s[9];
#pragma unroll
    for (int kh = 0; kh < 3; kh++)
#pragma unroll
        for (int kw = 0; kw < 3; kw++) s[kh * 3 + kw] = fsign(xp[kh * 54 + kw]);
    signed char* out = h1b + (size_t)idx * 32;
#pragma unroll 4
    for (int o = 0; o < 32; o++) {
        int acc = 0;
#pragma unroll
        for (int k = 0; k < 9; k++) acc += s[k] * (int)wb1[o * 9 + k];
        float inv = g1[o] * rsqrtf(var1[o] + 1e-5f);
        float z = ((float)acc + b1[o]) * inv + (be1[o] - mu1[o] * inv);
        out[o] = z > 0.f ? 1 : 0;
    }
}

__global__ void conv2_wmma_kernel(const signed char* __restrict__ h1b,
                                  const signed char* __restrict__ wb2,
                                  const float* __restrict__ b2, const float* __restrict__ g2,
                                  const float* __restrict__ be2, const float* __restrict__ mu2,
                                  const float* __restrict__ var2, signed char* __restrict__ h2b) {
    int wave = (blockIdx.x * blockDim.x + threadIdx.x) >> 5;
    int lane = threadIdx.x & 31;
    int mtile = wave;
    int col = lane & 15, hi = lane >> 4;

    int mA = mtile * 16 + col;
    int bA = mA / 60, r = mA % 60;
    int oh = r / 12, ow = r % 12;
    const signed char* abase =
        h1b + ((((size_t)bA * 11 + oh * 2) * 26 + ow * 2) << 5) + hi * 16;
    const signed char* brow = wb2 + col * 320 + hi * 16;

    v8i acc0 = {0, 0, 0, 0, 0, 0, 0, 0};
    v8i acc1 = {0, 0, 0, 0, 0, 0, 0, 0};
    v8i acc2 = {0, 0, 0, 0, 0, 0, 0, 0};
    v8i acc3 = {0, 0, 0, 0, 0, 0, 0, 0};
#pragma unroll
    for (int kc = 0; kc < 5; kc++) {
        AFrag A;
#pragma unroll
        for (int j = 0; j < 4; j++) {
            const int tap = kc * 2 + (j >> 1);
            int2 val = make_int2(0, 0);
            if (tap < 9) {
                const int kh = tap / 3, kw = tap % 3;
                const int off = (kh * 26 + kw) * 32 + (j & 1) * 8;
                val = *(const int2*)(abase + off);
            }
            A.d[j] = val;
        }
        BFrag B0, B1, B2, B3;
        B0.q[0] = *(const int4*)(brow + kc * 64);
        B0.q[1] = *(const int4*)(brow + kc * 64 + 32);
        B1.q[0] = *(const int4*)(brow + 16 * 320 + kc * 64);
        B1.q[1] = *(const int4*)(brow + 16 * 320 + kc * 64 + 32);
        B2.q[0] = *(const int4*)(brow + 32 * 320 + kc * 64);
        B2.q[1] = *(const int4*)(brow + 32 * 320 + kc * 64 + 32);
        B3.q[0] = *(const int4*)(brow + 48 * 320 + kc * 64);
        B3.q[1] = *(const int4*)(brow + 48 * 320 + kc * 64 + 32);
        acc0 = __builtin_amdgcn_wmma_i32_16x16x64_iu8(false, A.v, true, B0.v, acc0, false, false);
        acc1 = __builtin_amdgcn_wmma_i32_16x16x64_iu8(false, A.v, true, B1.v, acc1, false, false);
        acc2 = __builtin_amdgcn_wmma_i32_16x16x64_iu8(false, A.v, true, B2.v, acc2, false, false);
        acc3 = __builtin_amdgcn_wmma_i32_16x16x64_iu8(false, A.v, true, B3.v, acc3, false, false);
    }

    v8i accs[4] = {acc0, acc1, acc2, acc3};
#pragma unroll
    for (int nt = 0; nt < 4; nt++) {
        int n = nt * 16 + col;
        float inv = g2[n] * rsqrtf(var2[n] + 1e-5f);
        float sh = be2[n] - mu2[n] * inv;
        float bb = b2[n];
#pragma unroll
        for (int v = 0; v < 8; v++) {
            int m = mtile * 16 + v + hi * 8;
            int bI = m / 60, rr = m % 60;
            int ohh = rr / 12, oww = rr % 12;
            float z = ((float)accs[nt][v] + bb) * inv + sh;
            h2b[(size_t)bI * 3840 + n * 60 + ohh * 12 + oww] = z > 0.f ? 1 : 0;
        }
    }
}

__global__ void lin_wmma_kernel(const signed char* __restrict__ h2b,
                                const signed char* __restrict__ wbl,
                                const float* __restrict__ lin_b, signed char* __restrict__ h3b) {
    int wave = (blockIdx.x * blockDim.x + threadIdx.x) >> 5;
    int lane = threadIdx.x & 31;
    int npair = wave & 15;
    int mtile = wave >> 4;
    int col = lane & 15, hi = lane >> 4;

    int mA = mtile * 16 + col;
    int n0 = npair * 32 + col;
    int n1 = n0 + 16;
    int nc0 = n0 < 500 ? n0 : 499;
    int nc1 = n1 < 500 ? n1 : 499;
    const signed char* arow = h2b + (size_t)mA * 3840 + hi * 16;
    const signed char* brow0 = wbl + (size_t)nc0 * 3840 + hi * 16;
    const signed char* brow1 = wbl + (size_t)nc1 * 3840 + hi * 16;

    v8i acc0 = {0, 0, 0, 0, 0, 0, 0, 0};
    v8i acc1 = {0, 0, 0, 0, 0, 0, 0, 0};
    for (int kc = 0; kc < 60; kc++) {
        if (kc + 4 < 60) {
            __builtin_prefetch(arow + (kc + 4) * 64, 0, 1);
            __builtin_prefetch(brow0 + (kc + 4) * 64, 0, 1);
            __builtin_prefetch(brow1 + (kc + 4) * 64, 0, 1);
        }
        AFrag A; BFrag B0, B1;
        A.d[0] = *(const int2*)(arow + kc * 64);
        A.d[1] = *(const int2*)(arow + kc * 64 + 8);
        A.d[2] = *(const int2*)(arow + kc * 64 + 32);
        A.d[3] = *(const int2*)(arow + kc * 64 + 40);
        B0.q[0] = *(const int4*)(brow0 + kc * 64);
        B0.q[1] = *(const int4*)(brow0 + kc * 64 + 32);
        B1.q[0] = *(const int4*)(brow1 + kc * 64);
        B1.q[1] = *(const int4*)(brow1 + kc * 64 + 32);
        acc0 = __builtin_amdgcn_wmma_i32_16x16x64_iu8(false, A.v, true, B0.v, acc0, false, false);
        acc1 = __builtin_amdgcn_wmma_i32_16x16x64_iu8(false, A.v, true, B1.v, acc1, false, false);
    }

    if (n0 < 500) {
        float bb = lin_b[n0];
#pragma unroll
        for (int v = 0; v < 8; v++) {
            int m = mtile * 16 + v + hi * 8;
            h3b[(size_t)m * 500 + n0] = ((float)acc0[v] + bb) > 0.f ? 1 : 0;
        }
    }
    if (n1 < 500) {
        float bb = lin_b[n1];
#pragma unroll
        for (int v = 0; v < 8; v++) {
            int m = mtile * 16 + v + hi * 8;
            h3b[(size_t)m * 500 + n1] = ((float)acc1[v] + bb) > 0.f ? 1 : 0;
        }
    }
}

__global__ void out_kernel(const signed char* __restrict__ h3b, const signed char* __restrict__ wbo,
                           const float* __restrict__ out_b, float* __restrict__ y) {
    int idx = blockIdx.x * blockDim.x + threadIdx.x;
    if (idx >= 8192 * 10) return;
    int o = idx % 10, b = idx / 10;
    const signed char* h = h3b + (size_t)b * 500;
    const signed char* w = wbo + o * 500;
    int acc = 0;
#pragma unroll 4
    for (int k = 0; k < 500; k++) acc += (int)h[k] * (int)w[k];
    y[idx] = (float)acc + out_b[o];
}

extern "C" void kernel_launch(void* const* d_in, const int* in_sizes, int n_in,
                              void* d_out, int out_size, void* d_ws, size_t ws_size,
                              hipStream_t stream) {
    const float* x     = (const float*)d_in[0];
    const float* c1w   = (const float*)d_in[1];
    const float* c1b   = (const float*)d_in[2];
    const float* g1    = (const float*)d_in[3];
    const float* be1   = (const float*)d_in[4];
    const float* mu1   = (const float*)d_in[5];
    const float* var1  = (const float*)d_in[6];
    const float* c2w   = (const float*)d_in[7];
    const float* c2b   = (const float*)d_in[8];
    const float* g2    = (const float*)d_in[9];
    const float* be2   = (const float*)d_in[10];
    const float* mu2   = (const float*)d_in[11];
    const float* var2  = (const float*)d_in[12];
    const float* linw  = (const float*)d_in[13];
    const float* linb  = (const float*)d_in[14];
    const float* outw  = (const float*)d_in[15];
    const float* outb  = (const float*)d_in[16];
    float* y = (float*)d_out;

    char* ws = (char*)d_ws;
    signed char* wb1 = (signed char*)(ws + OFF_WB1);
    signed char* wb2 = (signed char*)(ws + OFF_WB2);
    signed char* wbo = (signed char*)(ws + OFF_WBO);
    signed char* wbl = (signed char*)(ws + OFF_WBL);
    signed char* h1b = (signed char*)(ws + OFF_H1B);
    signed char* h2b = (signed char*)(ws + OFF_H2B);
    signed char* h3b = (signed char*)(ws + OFF_H3B);

    {
        int total = 288 + 20480 + 5000 + 1920000;
        prep_kernel<<<(total + 255) / 256, 256, 0, stream>>>(c1w, c2w, linw, outw,
                                                             wb1, wb2, wbo, wbl);
    }
    {
        int total = 8192 * 11 * 26;
        conv1_kernel<<<(total + 255) / 256, 256, 0, stream>>>(x, wb1, c1b, g1, be1, mu1, var1, h1b);
    }
    conv2_wmma_kernel<<<30720 / 8, 256, 0, stream>>>(h1b, wb2, c2b, g2, be2, mu2, var2, h2b);
    lin_wmma_kernel<<<(512 * 16) / 8, 256, 0, stream>>>(h2b, wbl, linb, h3b);
    out_kernel<<<(8192 * 10 + 255) / 256, 256, 0, stream>>>(h3b, wbo, outb, y);
}
